// GemmaAttention_81750407512482
// MI455X (gfx1250) — matrix operand order repaired and checked by reading, not compiled, not run on hardware
//
#include <hip/hip_runtime.h>

typedef __bf16 bf16;
typedef __attribute__((ext_vector_type(8)))  __bf16 bf16x8;
typedef __attribute__((ext_vector_type(16))) __bf16 v16bf;
typedef __attribute__((ext_vector_type(8)))  float  v8f;

namespace cfg {
constexpr int  B   = 2;
constexpr int  S   = 2048;
constexpr int  D   = 2048;
constexpr int  H   = 8;
constexpr int  KVH = 1;
constexpr int  HD  = 256;
constexpr long QN  = (long)B * S * D;            // 8388608 (out elements)
}

// ---------------------------------------------------------------------------
// fp32 -> bf16 elementwise convert (weights / activations staging)
// ---------------------------------------------------------------------------
__global__ __launch_bounds__(256) void cvt_f32_bf16(const float* __restrict__ src,
                                                    bf16* __restrict__ dst, long n) {
  long i = ((long)blockIdx.x * blockDim.x + threadIdx.x) * 4;
  if (i + 3 < n) {
    float4 f = *(const float4*)(src + i);
    dst[i + 0] = (bf16)f.x; dst[i + 1] = (bf16)f.y;
    dst[i + 2] = (bf16)f.z; dst[i + 3] = (bf16)f.w;
  } else {
    for (; i < n; ++i) dst[i] = (bf16)src[i];
  }
}

// ---------------------------------------------------------------------------
// Generic 128x128-tile bf16 WMMA GEMM:  C[m,n] = scale * sum_k A[m,k]*B[n,k]
//   8 waves (2x4), each wave -> 64x32 patch = 4x2 wmma tiles of 16x16.
//   Double-buffered LDS (ping-pong, 40KB): fetch(t+1) -> compute(t) ->
//   commit(t+1) -> one barrier per k-step.
//   A may be fp32 in HBM (packed to bf16 in registers, b128 LDS commits).
//   MASKED adds the causal -1e9 term (n > m) after scaling (scores kernel).
//   batch offsets: blockIdx.z = zb*nh + zh  (b and head strides separate).
// ---------------------------------------------------------------------------
template <bool A_F32, bool OUT_F32, bool MASKED>
__global__ __launch_bounds__(256) void gemm_wmma_k(
    const void* __restrict__ Ap, long aSb, long aSh, int lda,
    const bf16* __restrict__ Bp, long bSb, long bSh, int ldb,
    void* __restrict__ Cp, long cSb, long cSh, int ldc,
    int K, int nh, float scale) {
  __shared__ __align__(16) bf16 tA[2][128][40];  // pad-40: b128-aligned, conflict-free
  __shared__ __align__(16) bf16 tB[2][128][40];

  const int tid  = threadIdx.x;
  const int lane = tid & 31;
  const int wave = tid >> 5;
  const int wr   = wave >> 2;      // 0..1 (row of 64)
  const int wc   = wave & 3;       // 0..3 (col of 32)
  const int zb   = blockIdx.z / nh;
  const int zh   = blockIdx.z % nh;
  const int m0   = blockIdx.y * 128;
  const int n0   = blockIdx.x * 128;

  const long aOff = (long)zb * aSb + (long)zh * aSh;
  const long bOff = (long)zb * bSb + (long)zh * bSh;
  const long cOff = (long)zb * cSb + (long)zh * cSh;

  const bf16*  Ab = A_F32 ? nullptr : ((const bf16*)Ap) + aOff;
  const float* Af = A_F32 ? ((const float*)Ap) + aOff : nullptr;
  const bf16*  Bb = Bp + bOff;

  const v8f vzero = {0.f, 0.f, 0.f, 0.f, 0.f, 0.f, 0.f, 0.f};
  v8f acc[4][2];
#pragma unroll
  for (int i = 0; i < 4; ++i) { acc[i][0] = vzero; acc[i][1] = vzero; }

  // global->LDS mapping: each thread moves 16 elems of A-tile and B-tile
  const int ldRow = tid >> 1;            // 0..127
  const int ldCol = (tid & 1) << 4;      // 0 or 16

  // fragment gather addressing (ISA 16-bit A / B layouts, wave32)
  const int fm    = lane & 15;           // M (A) or N (B) within tile
  const int base8 = (lane >> 4) << 3;    // A: K block {0,8}
  const int kb16  = (lane >> 4) << 4;    // unused; B also gathers with base8

  union FragU { v16bf v; bf16x8 h[2]; };

  // register staging for the in-flight tile
  float4 fA[4];
  bf16x8 rA[2], rB[2];

  auto fetch = [&](int kt) {
    const int k0 = kt * 32;
    if (A_F32) {
      const float* s = Af + (long)(m0 + ldRow) * lda + k0 + ldCol;
#pragma unroll
      for (int j = 0; j < 4; ++j) fA[j] = ((const float4*)s)[j];
    } else {
      const bf16* s = Ab + (long)(m0 + ldRow) * lda + k0 + ldCol;
      rA[0] = *(const bf16x8*)s;
      rA[1] = *(const bf16x8*)(s + 8);
    }
    const bf16* sb = Bb + (long)(n0 + ldRow) * ldb + k0 + ldCol;
    rB[0] = *(const bf16x8*)sb;
    rB[1] = *(const bf16x8*)(sb + 8);
  };

  auto commit = [&](int buf) {
    if (A_F32) {
      bf16x8 c0, c1;
      c0[0] = (bf16)fA[0].x; c0[1] = (bf16)fA[0].y; c0[2] = (bf16)fA[0].z; c0[3] = (bf16)fA[0].w;
      c0[4] = (bf16)fA[1].x; c0[5] = (bf16)fA[1].y; c0[6] = (bf16)fA[1].z; c0[7] = (bf16)fA[1].w;
      c1[0] = (bf16)fA[2].x; c1[1] = (bf16)fA[2].y; c1[2] = (bf16)fA[2].z; c1[3] = (bf16)fA[2].w;
      c1[4] = (bf16)fA[3].x; c1[5] = (bf16)fA[3].y; c1[6] = (bf16)fA[3].z; c1[7] = (bf16)fA[3].w;
      *(bf16x8*)&tA[buf][ldRow][ldCol]     = c0;
      *(bf16x8*)&tA[buf][ldRow][ldCol + 8] = c1;
    } else {
      *(bf16x8*)&tA[buf][ldRow][ldCol]     = rA[0];
      *(bf16x8*)&tA[buf][ldRow][ldCol + 8] = rA[1];
    }
    *(bf16x8*)&tB[buf][ldRow][ldCol]     = rB[0];
    *(bf16x8*)&tB[buf][ldRow][ldCol + 8] = rB[1];
  };

  auto compute = [&](int buf) {
    FragU fb[2];
#pragma unroll
    for (int ni = 0; ni < 2; ++ni) {
      const bf16* bp = &tB[buf][wc * 32 + ni * 16 + fm][base8];
      fb[ni].h[0] = *(const bf16x8*)bp;
      fb[ni].h[1] = *(const bf16x8*)(bp + 16);
    }
#pragma unroll
    for (int mi = 0; mi < 4; ++mi) {
      FragU fa;
      const bf16* ap = &tA[buf][wr * 64 + mi * 16 + fm][base8];
      fa.h[0] = *(const bf16x8*)ap;
      fa.h[1] = *(const bf16x8*)(ap + 16);
      acc[mi][0] = __builtin_amdgcn_wmma_f32_16x16x32_bf16(
          false, fa.v, false, fb[0].v, (short)0, acc[mi][0], false, false);
      acc[mi][1] = __builtin_amdgcn_wmma_f32_16x16x32_bf16(
          false, fa.v, false, fb[1].v, (short)0, acc[mi][1], false, false);
    }
  };

  const int nk = K >> 5;
  fetch(0);
  commit(0);
  __syncthreads();
  for (int t = 0; t < nk; ++t) {
    const int cur = t & 1;
    const bool more = (t + 1) < nk;
    if (more) fetch(t + 1);
    if (t + 2 < nk) {  // warm L2->WGP path for tile t+2 (global_prefetch_b8)
      const int kp = (t + 2) * 32;
      __builtin_prefetch(A_F32
          ? (const void*)(Af + (long)(m0 + ldRow) * lda + kp + ldCol)
          : (const void*)(Ab + (long)(m0 + ldRow) * lda + kp + ldCol), 0, 1);
      __builtin_prefetch((const void*)(Bb + (long)(n0 + ldRow) * ldb + kp + ldCol), 0, 1);
    }
    compute(cur);                 // WMMAs on buf `cur` while fetched regs land
    if (more) commit(cur ^ 1);    // reg -> LDS for tile t+1
    __syncthreads();              // one barrier per k-step
  }

  float* Cf = OUT_F32 ? ((float*)Cp) + cOff : nullptr;
  bf16*  Cb = OUT_F32 ? nullptr : ((bf16*)Cp) + cOff;
  const int mlane = (lane >> 4) << 3;    // C layout: M = r + 8*(lane/16)
#pragma unroll
  for (int mi = 0; mi < 4; ++mi) {
#pragma unroll
    for (int ni = 0; ni < 2; ++ni) {
      const int gn = n0 + wc * 32 + ni * 16 + fm;
#pragma unroll
      for (int r = 0; r < 8; ++r) {
        const int gm = m0 + wr * 64 + mi * 16 + mlane + r;
        float v = acc[mi][ni][r] * scale;
        if (MASKED) v += (gn <= gm) ? 0.0f : -1e9f;
        if (OUT_F32) Cf[(long)gm * ldc + gn] = v;
        else         Cb[(long)gm * ldc + gn] = (bf16)v;
      }
    }
  }
}

// ---------------------------------------------------------------------------
// RoPE in place on bf16 Q [B*S, H*HD] and K [B*S, KVH*HD].
// Thread -> (bs, hh in [0,H+KVH), i in [0,HD/2)); hh==H selects K head.
// ---------------------------------------------------------------------------
__global__ __launch_bounds__(256) void rope_k(bf16* __restrict__ Q, bf16* __restrict__ Kc,
                                              const int* __restrict__ pos) {
  using namespace cfg;
  const long idx = (long)blockIdx.x * blockDim.x + threadIdx.x;
  const int  i   = (int)(idx & (HD / 2 - 1));          // 0..127
  const long t   = idx >> 7;
  const int  hh  = (int)(t % (H + KVH));
  const long bs  = t / (H + KVH);                      // 0..B*S-1
  const float p  = (float)pos[bs];
  // inv_freq = 10000^(-2i/HD) = exp(-(2i/HD)*ln(10000))
  const float invf = __expf(-(float)(2 * i) * (9.210340371976184f / (float)HD));
  const float f = p * invf;
  const float c = __cosf(f), sn = __sinf(f);
  bf16* base = (hh < H) ? (Q + bs * (long)(H * HD) + (long)hh * HD)
                        : (Kc + bs * (long)(KVH * HD));
  const float x1 = (float)base[i];
  const float x2 = (float)base[i + HD / 2];
  base[i]          = (bf16)(x1 * c - x2 * sn);
  base[i + HD / 2] = (bf16)(x2 * c + x1 * sn);
}

// ---------------------------------------------------------------------------
// Transpose V[b][s][HD] -> Vt[b][HD][s] so the ctx GEMM streams k contiguously.
// ---------------------------------------------------------------------------
__global__ __launch_bounds__(256) void vtrans_k(const bf16* __restrict__ V,
                                                bf16* __restrict__ Vt) {
  using namespace cfg;
  const long i = (long)blockIdx.x * blockDim.x + threadIdx.x;   // over B*S*HD
  const int  n = (int)(i & (HD - 1));
  const long t = i >> 8;
  const int  s = (int)(t % S);
  const long b = t / S;
  Vt[(b * HD + n) * (long)S + s] = V[i];
}

// ---------------------------------------------------------------------------
// In-place fp32 row softmax over attn rows of length S=2048 (block per row).
// ---------------------------------------------------------------------------
__global__ __launch_bounds__(256) void softmax_k(float* __restrict__ attn) {
  __shared__ float red[256];
  const long  row = blockIdx.x;
  float* p = attn + row * (long)cfg::S;
  const int tid = threadIdx.x;
  float4 a = ((float4*)p)[tid * 2];
  float4 b = ((float4*)p)[tid * 2 + 1];
  float m = fmaxf(fmaxf(fmaxf(a.x, a.y), fmaxf(a.z, a.w)),
                  fmaxf(fmaxf(b.x, b.y), fmaxf(b.z, b.w)));
  red[tid] = m; __syncthreads();
  for (int s = 128; s > 0; s >>= 1) {
    if (tid < s) red[tid] = fmaxf(red[tid], red[tid + s]);
    __syncthreads();
  }
  const float rmax = red[0]; __syncthreads();
  float4 ea, eb;
  ea.x = __expf(a.x - rmax); ea.y = __expf(a.y - rmax);
  ea.z = __expf(a.z - rmax); ea.w = __expf(a.w - rmax);
  eb.x = __expf(b.x - rmax); eb.y = __expf(b.y - rmax);
  eb.z = __expf(b.z - rmax); eb.w = __expf(b.w - rmax);
  red[tid] = ea.x + ea.y + ea.z + ea.w + eb.x + eb.y + eb.z + eb.w;
  __syncthreads();
  for (int s = 128; s > 0; s >>= 1) {
    if (tid < s) red[tid] += red[tid + s];
    __syncthreads();
  }
  const float inv = 1.0f / red[0];
  ea.x *= inv; ea.y *= inv; ea.z *= inv; ea.w *= inv;
  eb.x *= inv; eb.y *= inv; eb.z *= inv; eb.w *= inv;
  ((float4*)p)[tid * 2]     = ea;
  ((float4*)p)[tid * 2 + 1] = eb;
}

// ---------------------------------------------------------------------------
extern "C" void kernel_launch(void* const* d_in, const int* in_sizes, int n_in,
                              void* d_out, int out_size, void* d_ws, size_t ws_size,
                              hipStream_t stream) {
  using namespace cfg;
  const float* X   = (const float*)d_in[0];
  const int*   pos = (const int*)d_in[2];
  const float* Wq  = (const float*)d_in[3];
  const float* Wk  = (const float*)d_in[4];
  const float* Wv  = (const float*)d_in[5];
  const float* Wo  = (const float*)d_in[6];

  float* out  = (float*)d_out;
  float* attn = out + QN;               // out [B,S,D] then attn [B,H,S,S]

  // workspace carve-out (bf16 staging, ~72 MB total)
  const long nX  = (long)B * S * D;         // 8388608
  const long nWq = (long)H * HD * D;        // 4194304
  const long nWk = (long)KVH * HD * D;      // 524288
  const long nWo = (long)D * H * HD;        // 4194304
  const long nQ  = (long)B * S * H * HD;    // 8388608
  const long nK  = (long)B * S * KVH * HD;  // 1048576
  char* ws = (char*)d_ws;
  bf16* Xb   = (bf16*)ws; ws += nX  * 2;
  bf16* Wqb  = (bf16*)ws; ws += nWq * 2;
  bf16* Wkb  = (bf16*)ws; ws += nWk * 2;
  bf16* Wvb  = (bf16*)ws; ws += nWk * 2;
  bf16* Wob  = (bf16*)ws; ws += nWo * 2;
  bf16* Qb   = (bf16*)ws; ws += nQ  * 2;
  bf16* Kb   = (bf16*)ws; ws += nK  * 2;
  bf16* Vb   = (bf16*)ws; ws += nK  * 2;
  bf16* Vtb  = (bf16*)ws; ws += nK  * 2;
  bf16* Ctxb = (bf16*)ws; ws += nQ  * 2;

  const dim3 blk(256);
  auto cvt = [&](const float* s, bf16* d, long n) {
    cvt_f32_bf16<<<dim3((unsigned)((n / 4 + 255) / 256)), blk, 0, stream>>>(s, d, n);
  };

  // 1. fp32 -> bf16 staging
  cvt(X, Xb, nX);
  cvt(Wq, Wqb, nWq);
  cvt(Wk, Wkb, nWk);
  cvt(Wv, Wvb, nWk);
  cvt(Wo, Wob, nWo);

  // 2. projections: C[m,n] = sum_k X[m,k] * W[n,k]
  gemm_wmma_k<false, false, false><<<dim3((H * HD) / 128, (B * S) / 128, 1), blk, 0, stream>>>(
      Xb, 0, 0, D, Wqb, 0, 0, D, Qb, 0, 0, H * HD, D, 1, 1.0f);
  gemm_wmma_k<false, false, false><<<dim3((KVH * HD) / 128, (B * S) / 128, 1), blk, 0, stream>>>(
      Xb, 0, 0, D, Wkb, 0, 0, D, Kb, 0, 0, KVH * HD, D, 1, 1.0f);
  gemm_wmma_k<false, false, false><<<dim3((KVH * HD) / 128, (B * S) / 128, 1), blk, 0, stream>>>(
      Xb, 0, 0, D, Wvb, 0, 0, D, Vb, 0, 0, KVH * HD, D, 1, 1.0f);

  // 3. RoPE on Q and K (in place)
  {
    const long nThreads = (long)B * S * (H + KVH) * (HD / 2);
    rope_k<<<dim3((unsigned)(nThreads / 256)), blk, 0, stream>>>(Qb, Kb, pos);
  }

  // 4. V transpose for ctx GEMM
  vtrans_k<<<dim3((unsigned)(nK / 256)), blk, 0, stream>>>(Vb, Vtb);

  // 5. scores = (1/16) Q K^T + causal mask -> attn region of d_out (fp32)
  gemm_wmma_k<false, true, true><<<dim3(S / 128, S / 128, B * H), blk, 0, stream>>>(
      Qb, (long)S * H * HD, HD, H * HD,
      Kb, (long)S * KVH * HD, 0, KVH * HD,
      attn, (long)H * S * S, (long)S * S, S,
      HD, H, 0.0625f);

  // 6. softmax rows, in place
  softmax_k<<<dim3((unsigned)((long)B * H * S)), blk, 0, stream>>>(attn);

  // 7. ctx = attn @ V  (A fp32 packed to bf16 in regs; B = Vt so k contiguous)
  gemm_wmma_k<true, false, false><<<dim3(HD / 128, S / 128, B * H), blk, 0, stream>>>(
      attn, (long)H * S * S, (long)S * S, S,
      Vtb, (long)HD * S, 0, S,
      Ctxb, (long)S * H * HD, HD, H * HD,
      S, H, 1.0f);

  // 8. out = ctx @ Wo^T (fp32 to d_out)
  gemm_wmma_k<false, true, false><<<dim3(D / 128, (B * S) / 128, 1), blk, 0, stream>>>(
      Ctxb, 0, 0, H * HD, Wob, 0, 0, H * HD, out, 0, 0, D, H * HD, 1, 1.0f);
}
